// AttentionDist_45947560132766
// MI455X (gfx1250) — matrix operand order repaired and checked by reading, not compiled, not run on hardware
//
#include <hip/hip_runtime.h>

#define BB 4
#define NN 2048
#define DD 1024
#define HH 16
#define KK 16
#define DHH 64
#define MM (BB * NN) /* 8192 */

typedef __attribute__((ext_vector_type(16))) __bf16 v16bf;
typedef __attribute__((ext_vector_type(8)))  __bf16 v8bf;
typedef __attribute__((ext_vector_type(8)))  float  v8f;

__device__ __forceinline__ unsigned short f2bf(float f) {
  union { float f; unsigned u; } v; v.f = f;
  unsigned u = v.u;
  u += 0x7FFFu + ((u >> 16) & 1u);   // round-to-nearest-even
  return (unsigned short)(u >> 16);
}

// ---------------- LayerNorm (fp32 in -> bf16 out), one row per block ----------
__global__ void __launch_bounds__(256)
ln_bf16_kernel(const float* __restrict__ x, const float* __restrict__ g,
               const float* __restrict__ be, unsigned short* __restrict__ xn) {
  int row = blockIdx.x;                       // 0..MM-1
  const float* xr = x + (size_t)row * DD;
  __shared__ float s1[256], s2[256];
  float a = 0.f, b2 = 0.f;
  for (int i = threadIdx.x; i < DD; i += 256) { float t = xr[i]; a += t; b2 += t * t; }
  s1[threadIdx.x] = a; s2[threadIdx.x] = b2;
  __syncthreads();
  for (int s = 128; s > 0; s >>= 1) {
    if ((int)threadIdx.x < s) { s1[threadIdx.x] += s1[threadIdx.x + s]; s2[threadIdx.x] += s2[threadIdx.x + s]; }
    __syncthreads();
  }
  float mu  = s1[0] * (1.0f / DD);
  float var = s2[0] * (1.0f / DD) - mu * mu;
  float rs  = rsqrtf(var + 1e-5f);
  for (int i = threadIdx.x; i < DD; i += 256) {
    float t = (xr[i] - mu) * rs * g[i] + be[i];
    xn[(size_t)row * DD + i] = f2bf(t);
  }
}

// ---------- Weight transpose + bf16 convert: Wt[n,k] = W[k,n] -----------------
__global__ void __launch_bounds__(256)
transp_bf16_kernel(const float* __restrict__ w, unsigned short* __restrict__ wt) {
  __shared__ float tile[32][33];
  int k0 = blockIdx.y * 32;                   // source row
  int n0 = blockIdx.x * 32;                   // source col
  for (int r = threadIdx.y; r < 32; r += 8)
    tile[r][threadIdx.x] = w[(size_t)(k0 + r) * DD + n0 + threadIdx.x];
  __syncthreads();
  for (int r = threadIdx.y; r < 32; r += 8)
    wt[(size_t)(n0 + r) * DD + k0 + threadIdx.x] = f2bf(tile[threadIdx.x][r]);
}

// ------- bf16 WMMA GEMM: 16x64 tile per wave, double-buffered K-loop ----------
// C[M,1024] = A[M,1024] x W ; B fed from pre-transposed Wt[n,k].
__device__ __forceinline__ v16bf load_a_frag(const unsigned short* __restrict__ pa, int k0) {
  v8bf a0 = *(const v8bf*)(pa + k0);          // K = k0 + {0..7}   (hi lanes: +16)
  v8bf a1 = *(const v8bf*)(pa + k0 + 8);      // K = k0 + {8..15}  (hi lanes: +16)
  v16bf a;
#pragma unroll
  for (int i = 0; i < 8; ++i) { a[i] = a0[i]; a[8 + i] = a1[i]; }
  return a;
}

__global__ void __launch_bounds__(256)
wmma_gemm_bf16_kernel(const unsigned short* __restrict__ Abf,
                      const unsigned short* __restrict__ Bt,
                      float* __restrict__ C) {
  int lane = threadIdx.x & 31;
  int wave = threadIdx.x >> 5;                // 0..7 -> 8 x 16 = 128 rows/block
  int m0 = blockIdx.y * 128 + wave * 16;
  int n0 = blockIdx.x * 64;
  int rowA = m0 + (lane & 15);
  int colB = n0 + (lane & 15);
  const unsigned short* pa = Abf + (size_t)rowA * DD + (lane < 16 ? 0 : 16);
  const unsigned short* pb = Bt  + (size_t)colB * DD + (lane < 16 ? 0 : 16);

  v8f acc0 = {}, acc1 = {}, acc2 = {}, acc3 = {};

  // prologue: fragments for k0 = 0
  v16bf a  = load_a_frag(pa, 0);
  v16bf b0 = *(const v16bf*)(pb);
  v16bf b1 = *(const v16bf*)(pb + (size_t)16 * DD);
  v16bf b2 = *(const v16bf*)(pb + (size_t)32 * DD);
  v16bf b3 = *(const v16bf*)(pb + (size_t)48 * DD);

  for (int k0 = 0; k0 < DD - 32; k0 += 32) {
    int kn = k0 + 32;
    __builtin_prefetch(pa + kn + 128, 0, 1);   // global_prefetch_b8
    // issue next-step loads before consuming current fragments
    v16bf an  = load_a_frag(pa, kn);
    v16bf bn0 = *(const v16bf*)(pb + kn);
    v16bf bn1 = *(const v16bf*)(pb + (size_t)16 * DD + kn);
    v16bf bn2 = *(const v16bf*)(pb + (size_t)32 * DD + kn);
    v16bf bn3 = *(const v16bf*)(pb + (size_t)48 * DD + kn);

    acc0 = __builtin_amdgcn_wmma_f32_16x16x32_bf16(false, a, false, b0, (short)0, acc0, false, false);
    acc1 = __builtin_amdgcn_wmma_f32_16x16x32_bf16(false, a, false, b1, (short)0, acc1, false, false);
    acc2 = __builtin_amdgcn_wmma_f32_16x16x32_bf16(false, a, false, b2, (short)0, acc2, false, false);
    acc3 = __builtin_amdgcn_wmma_f32_16x16x32_bf16(false, a, false, b3, (short)0, acc3, false, false);

    a = an; b0 = bn0; b1 = bn1; b2 = bn2; b3 = bn3;
  }
  // epilogue: last K-step
  acc0 = __builtin_amdgcn_wmma_f32_16x16x32_bf16(false, a, false, b0, (short)0, acc0, false, false);
  acc1 = __builtin_amdgcn_wmma_f32_16x16x32_bf16(false, a, false, b1, (short)0, acc1, false, false);
  acc2 = __builtin_amdgcn_wmma_f32_16x16x32_bf16(false, a, false, b2, (short)0, acc2, false, false);
  acc3 = __builtin_amdgcn_wmma_f32_16x16x32_bf16(false, a, false, b3, (short)0, acc3, false, false);

  int rb = m0 + (lane < 16 ? 0 : 8);
  int cc = lane & 15;
#pragma unroll
  for (int j = 0; j < 8; ++j) {
    float* crow = C + (size_t)(rb + j) * DD + n0 + cc;
    crow[0]  = acc0[j];
    crow[16] = acc1[j];
    crow[32] = acc2[j];
    crow[48] = acc3[j];
  }
}

// ---------------- top-16 nearest valid neighbors, one thread per query --------
__global__ void __launch_bounds__(256)
topk_kernel(const float* __restrict__ coord, const unsigned char* __restrict__ mask,
            int* __restrict__ idxo, float* __restrict__ dno) {
  int q = blockIdx.x * blockDim.x + threadIdx.x;
  if (q >= MM) return;
  int b = q / NN, n = q % NN;
  const float* cb = coord + (size_t)b * NN * 2;
  float cx = cb[2 * n], cy = cb[2 * n + 1];
  const unsigned char* mrow = mask + ((size_t)b * NN + n) * NN;
  float bd[KK]; int bi[KK];
#pragma unroll
  for (int t = 0; t < KK; ++t) { bd[t] = 3.4e38f; bi[t] = -1; }
  for (int j = 0; j < NN; ++j) {
    if (!mrow[j]) continue;
    float dx = cx - cb[2 * j], dy = cy - cb[2 * j + 1];
    float d = sqrtf(dx * dx + dy * dy + 1e-12f);
    if (d < bd[KK - 1]) {
      float cd = d; int ci = j;
#pragma unroll
      for (int t = 0; t < KK; ++t) {
        if (cd < bd[t]) { float td = bd[t]; int ti = bi[t]; bd[t] = cd; bi[t] = ci; cd = td; ci = ti; }
      }
    }
  }
#pragma unroll
  for (int t = 0; t < KK; ++t) {
    idxo[(size_t)q * KK + t] = bi[t];
    dno [(size_t)q * KK + t] = bi[t] >= 0 ? bd[t] : 0.0f;
  }
}

// ------ neighbor attention: scores + softmax + weighted sum, bf16 output ------
__global__ void __launch_bounds__(256)
attn_kernel(const float* __restrict__ Q, const float* __restrict__ Km,
            const float* __restrict__ Vm, const int* __restrict__ idx,
            const float* __restrict__ dn, const float* __restrict__ relw,
            const float* __restrict__ relb, unsigned short* __restrict__ O) {
  int t = blockIdx.x * blockDim.x + threadIdx.x;   // over B*H*N
  if (t >= BB * HH * NN) return;
  int n  = t % NN;
  int bh = t / NN;
  int h  = bh % HH;
  int b  = bh / HH;
  int row = b * NN + n;
  int off = h * DHH;
  const float* qv = Q + (size_t)row * DD + off;
  float rw = relw[h], rb = relb[h];
  float sc[KK]; int nid[KK];
#pragma unroll
  for (int kk = 0; kk < KK; ++kk) {
    int j = idx[(size_t)row * KK + kk];
    nid[kk] = j;
    float s = -3.0e38f;
    if (j >= 0) {
      const float* kv = Km + ((size_t)b * NN + j) * DD + off;
      float acc = 0.f;
      for (int d2 = 0; d2 < DHH; ++d2) acc = fmaf(qv[d2], kv[d2], acc);
      s = acc * 0.125f + rw * dn[(size_t)row * KK + kk] + rb;  // 1/sqrt(64)=0.125
    }
    sc[kk] = s;
  }
  float mx = sc[0];
#pragma unroll
  for (int kk = 1; kk < KK; ++kk) mx = fmaxf(mx, sc[kk]);
  float sum = 0.f;
#pragma unroll
  for (int kk = 0; kk < KK; ++kk) { sc[kk] = __expf(sc[kk] - mx); sum += sc[kk]; }
  float inv = 1.0f / sum;
  for (int c = 0; c < DHH; c += 16) {
    float acc16[16];
#pragma unroll
    for (int u = 0; u < 16; ++u) acc16[u] = 0.f;
#pragma unroll
    for (int kk = 0; kk < KK; ++kk) {
      int j = nid[kk]; if (j < 0) continue;
      float w = sc[kk] * inv;
      const float* vv = Vm + ((size_t)b * NN + j) * DD + off + c;
#pragma unroll
      for (int u = 0; u < 16; ++u) acc16[u] = fmaf(w, vv[u], acc16[u]);
    }
#pragma unroll
    for (int u = 0; u < 16; ++u) O[(size_t)row * DD + off + c + u] = f2bf(acc16[u]);
  }
}

extern "C" void kernel_launch(void* const* d_in, const int* in_sizes, int n_in,
                              void* d_out, int out_size, void* d_ws, size_t ws_size,
                              hipStream_t stream) {
  (void)in_sizes; (void)n_in; (void)out_size; (void)ws_size;
  const float*         x     = (const float*)d_in[0];
  const float*         coord = (const float*)d_in[1];
  const unsigned char* mask  = (const unsigned char*)d_in[2];
  const float*         g     = (const float*)d_in[3];
  const float*         be    = (const float*)d_in[4];
  const float*         Wq    = (const float*)d_in[5];
  const float*         Wk    = (const float*)d_in[6];
  const float*         Wv    = (const float*)d_in[7];
  const float*         Wo    = (const float*)d_in[8];
  const float*         relw  = (const float*)d_in[9];
  const float*         relb  = (const float*)d_in[10];
  float* out = (float*)d_out;

  char* ws = (char*)d_ws;
  size_t o = 0;
  auto alloc = [&](size_t bytes) -> void* {
    o = (o + 255) & ~(size_t)255;
    void* p = ws + o;
    o += bytes;
    return p;
  };
  unsigned short* xn  = (unsigned short*)alloc((size_t)MM * DD * 2);
  unsigned short* wtq = (unsigned short*)alloc((size_t)DD * DD * 2);
  unsigned short* wtk = (unsigned short*)alloc((size_t)DD * DD * 2);
  unsigned short* wtv = (unsigned short*)alloc((size_t)DD * DD * 2);
  unsigned short* wto = (unsigned short*)alloc((size_t)DD * DD * 2);
  float*          qf  = (float*)alloc((size_t)MM * DD * 4);
  float*          kf  = (float*)alloc((size_t)MM * DD * 4);
  float*          vf  = (float*)alloc((size_t)MM * DD * 4);
  unsigned short* ao  = (unsigned short*)alloc((size_t)MM * DD * 2);
  int*            idx = (int*)alloc((size_t)MM * KK * 4);
  float*          dnb = (float*)alloc((size_t)MM * KK * 4);

  // 1) LayerNorm -> bf16 activations
  ln_bf16_kernel<<<MM, 256, 0, stream>>>(x, g, be, xn);

  // 2) Transpose+convert the four weight matrices
  dim3 tb(32, 8), tg(DD / 32, DD / 32);
  transp_bf16_kernel<<<tg, tb, 0, stream>>>(Wq, wtq);
  transp_bf16_kernel<<<tg, tb, 0, stream>>>(Wk, wtk);
  transp_bf16_kernel<<<tg, tb, 0, stream>>>(Wv, wtv);
  transp_bf16_kernel<<<tg, tb, 0, stream>>>(Wo, wto);

  // 3) Q/K/V projections via WMMA bf16 (16x64 tile per wave, 128x64 per block)
  dim3 gb(256), gg(DD / 64, MM / 128);
  wmma_gemm_bf16_kernel<<<gg, gb, 0, stream>>>(xn, wtq, qf);
  wmma_gemm_bf16_kernel<<<gg, gb, 0, stream>>>(xn, wtk, kf);
  wmma_gemm_bf16_kernel<<<gg, gb, 0, stream>>>(xn, wtv, vf);

  // 4) distances + top-16 neighbors
  topk_kernel<<<MM / 256, 256, 0, stream>>>(coord, mask, idx, dnb);

  // 5) neighbor attention -> bf16 heads output
  attn_kernel<<<(BB * HH * NN) / 256, 256, 0, stream>>>(qf, kf, vf, idx, dnb, relw, relb, ao);

  // 6) output projection via WMMA bf16 -> fp32 result
  wmma_gemm_bf16_kernel<<<gg, gb, 0, stream>>>(ao, wto, out);
}
